// GraphSAGE_68848325755000
// MI455X (gfx1250) — matrix operand order repaired and checked by reading, not compiled, not run on hardware
//
#include <hip/hip_runtime.h>
#include <hip/hip_bf16.h>

#define NN 8192
#define FF 256
#define CC 64
#define KN1 25
#define KN2 10

typedef __attribute__((ext_vector_type(16))) __bf16 v16bf;
typedef __attribute__((ext_vector_type(8)))  float  v8f;

union Frag32B { v16bf v; uint4 q[2]; };

__device__ __forceinline__ float leaky(float x) { return x > 0.0f ? x : 0.01f * x; }

// ---------------------------------------------------------------------------
// Kernel 1: per-row first-k neighbor selection via wave32 ballot scan.
// One wave per adjacency row; early exit once 25 neighbors found (density~0.5
// -> ~2 iterations), avoiding the full 256MB adjacency read.
// ---------------------------------------------------------------------------
__global__ __launch_bounds__(32) void select_first_k_kernel(
    const int* __restrict__ A, int* __restrict__ idx,
    int* __restrict__ cnt1, int* __restrict__ cnt2)
{
    const int row  = blockIdx.x;
    const int lane = threadIdx.x;                 // 0..31 (wave32)
    const int* arow = A + (size_t)row * NN;
    int total = 0;
    for (int base = 0; base < NN && total < KN1; base += 32) {
        __builtin_prefetch(arow + base + 64, 0, 1);   // global_prefetch_b8
        const int v = arow[base + lane];
        const unsigned mask = (unsigned)__ballot(v > 0);   // wave32: low 32 bits
        if (v > 0) {
            const int r = total + __popc(mask & ((1u << lane) - 1u));
            if (r < KN1) idx[(size_t)row * KN1 + r] = base + lane;
        }
        total += __popc(mask);
    }
    if (lane == 0) {
        cnt1[row] = total < KN1 ? total : KN1;
        cnt2[row] = total < KN2 ? total : KN2;
    }
}

// ---------------------------------------------------------------------------
// Kernel 2: fp32 -> bf16 conversion (grid-stride).
// ---------------------------------------------------------------------------
__global__ void f32_to_bf16_kernel(const float* __restrict__ in,
                                   __bf16* __restrict__ out, int n)
{
    int i = blockIdx.x * blockDim.x + threadIdx.x;
    const int stride = gridDim.x * blockDim.x;
    for (; i < n; i += stride) out[i] = (__bf16)in[i];
}

// ---------------------------------------------------------------------------
// Kernel 3: neighbor mean via index gather. One block (256 threads) per row;
// thread t owns column t -> fully coalesced row gathers. fp32 accumulate.
// ---------------------------------------------------------------------------
__global__ __launch_bounds__(FF) void aggregate_mean_kernel(
    const __bf16* __restrict__ Xb, const int* __restrict__ idx,
    const int* __restrict__ cnt, __bf16* __restrict__ Mout)
{
    const int row = blockIdx.x;
    const int col = threadIdx.x;
    const int c = cnt[row];
    float s = 0.0f;
    for (int j = 0; j < c; ++j) {
        const int nb = idx[(size_t)row * KN1 + j];
        s += (float)Xb[(size_t)nb * FF + col];
    }
    s *= 1.0f / (float)(c > 0 ? c : 1);
    Mout[(size_t)row * FF + col] = (__bf16)s;
}

// ---------------------------------------------------------------------------
// Kernel 4: fused dual GEMM + bias + leaky-ReLU + neighbor gating.
//   out = leaky(X @ Wl^T + bl) + (cnt>0) * leaky(M @ Wn^T + bn)
// Tile: 16 rows x 256 cols per 256-thread block (8 waves); each wave owns two
// 16x16 column tiles. K loop = 8 x v_wmma_f32_16x16x32_bf16 (4 wmma/iter).
// A-frag layout (16-bit A 16x32, ISA 7.12.2): lane L holds row L%16,
//   K = k0 + {0..7,16..23} (lanes 0-15) or k0 + {8..15,24..31} (lanes 16-31).
// B-frag (32x16): lane L holds column L%16 (= weight row), same K order as A:
//   K = k0 + {0..7,16..23} (lanes 0-15) or k0 + {8..15,24..31} (lanes 16-31).
// ---------------------------------------------------------------------------
__global__ __launch_bounds__(256) void dual_gemm_leaky_kernel(
    const __bf16* __restrict__ Xin, const __bf16* __restrict__ Min,
    const __bf16* __restrict__ Wl, const float* __restrict__ bl,
    const __bf16* __restrict__ Wn, const float* __restrict__ bn,
    const int* __restrict__ cnt, __bf16* __restrict__ Hout)
{
    const int K = FF, H = FF;
    const int tid   = threadIdx.x;
    const int wave  = tid >> 5;
    const int lane  = tid & 31;
    const int l16   = lane & 15;
    const int lhalf = lane >> 4;
    const int rowBase = blockIdx.x * 16;
    const int ct0 = wave * 2, ct1 = wave * 2 + 1;

    v8f aL0 = {}, aL1 = {}, aN0 = {}, aN1 = {};
    const size_t arow = (size_t)(rowBase + l16) * K;
    const size_t w0   = (size_t)(ct0 * 16 + l16) * K;
    const size_t w1   = (size_t)(ct1 * 16 + l16) * K;

    for (int k0 = 0; k0 < K; k0 += 32) {
        Frag32B ax, am, fl0, fl1, fn0, fn1;
        ax.q[0] = *(const uint4*)(Xin + arow + k0 + lhalf * 8);
        ax.q[1] = *(const uint4*)(Xin + arow + k0 + 16 + lhalf * 8);
        am.q[0] = *(const uint4*)(Min + arow + k0 + lhalf * 8);
        am.q[1] = *(const uint4*)(Min + arow + k0 + 16 + lhalf * 8);

        fl0.q[0] = *(const uint4*)(Wl + w0 + k0 + lhalf * 8);
        fl0.q[1] = *(const uint4*)(Wl + w0 + k0 + 16 + lhalf * 8);
        fl1.q[0] = *(const uint4*)(Wl + w1 + k0 + lhalf * 8);
        fl1.q[1] = *(const uint4*)(Wl + w1 + k0 + 16 + lhalf * 8);
        fn0.q[0] = *(const uint4*)(Wn + w0 + k0 + lhalf * 8);
        fn0.q[1] = *(const uint4*)(Wn + w0 + k0 + 16 + lhalf * 8);
        fn1.q[0] = *(const uint4*)(Wn + w1 + k0 + lhalf * 8);
        fn1.q[1] = *(const uint4*)(Wn + w1 + k0 + 16 + lhalf * 8);

        aL0 = __builtin_amdgcn_wmma_f32_16x16x32_bf16(false, ax.v, false, fl0.v, (short)0, aL0, false, false);
        aL1 = __builtin_amdgcn_wmma_f32_16x16x32_bf16(false, ax.v, false, fl1.v, (short)0, aL1, false, false);
        aN0 = __builtin_amdgcn_wmma_f32_16x16x32_bf16(false, am.v, false, fn0.v, (short)0, aN0, false, false);
        aN1 = __builtin_amdgcn_wmma_f32_16x16x32_bf16(false, am.v, false, fn1.v, (short)0, aN1, false, false);
    }

    // Epilogue. C/D layout (32-bit 16x16): lane L holds col L%16; VGPR r holds
    // row r (lanes 0-15) or 8+r (lanes 16-31).
    const int col0 = ct0 * 16 + l16;
    const int col1 = ct1 * 16 + l16;
    const float bL0 = bl[col0], bL1 = bl[col1];
    const float bN0 = bn[col0], bN1 = bn[col1];
    #pragma unroll
    for (int r = 0; r < 8; ++r) {
        const int row = rowBase + lhalf * 8 + r;
        const float g = (cnt[row] > 0) ? 1.0f : 0.0f;
        const float o0 = leaky(aL0[r] + bL0) + g * leaky(aN0[r] + bN0);
        const float o1 = leaky(aL1[r] + bL1) + g * leaky(aN1[r] + bN1);
        Hout[(size_t)row * H + col0] = (__bf16)o0;
        Hout[(size_t)row * H + col1] = (__bf16)o1;
    }
}

// ---------------------------------------------------------------------------
// Kernel 5: logits = H2 @ W3^T + b3. 16x64 tile per 128-thread block; one
// 16x16 column tile per wave. fp32 logits out.
// ---------------------------------------------------------------------------
__global__ __launch_bounds__(128) void final_gemm_kernel(
    const __bf16* __restrict__ Hin, const __bf16* __restrict__ W3b,
    const float* __restrict__ b3, float* __restrict__ logits)
{
    const int K = FF;
    const int tid   = threadIdx.x;
    const int wave  = tid >> 5;                // column tile 0..3
    const int lane  = tid & 31;
    const int l16   = lane & 15;
    const int lhalf = lane >> 4;
    const int rowBase = blockIdx.x * 16;

    v8f acc = {};
    const size_t arow = (size_t)(rowBase + l16) * K;
    const size_t wrow = (size_t)(wave * 16 + l16) * K;

    for (int k0 = 0; k0 < K; k0 += 32) {
        Frag32B a, b;
        a.q[0] = *(const uint4*)(Hin + arow + k0 + lhalf * 8);
        a.q[1] = *(const uint4*)(Hin + arow + k0 + 16 + lhalf * 8);
        b.q[0] = *(const uint4*)(W3b + wrow + k0 + lhalf * 8);
        b.q[1] = *(const uint4*)(W3b + wrow + k0 + 16 + lhalf * 8);
        acc = __builtin_amdgcn_wmma_f32_16x16x32_bf16(false, a.v, false, b.v, (short)0, acc, false, false);
    }

    const int col = wave * 16 + l16;
    const float bias = b3[col];
    #pragma unroll
    for (int r = 0; r < 8; ++r) {
        const int row = rowBase + lhalf * 8 + r;
        logits[(size_t)row * CC + col] = acc[r] + bias;
    }
}

// ---------------------------------------------------------------------------
// Kernel 6: row-wise log_softmax over 64 logits. One wave per row, 2 elements
// per lane, shuffle reductions.
// ---------------------------------------------------------------------------
__global__ __launch_bounds__(32) void log_softmax64_kernel(
    const float* __restrict__ logits, float* __restrict__ out)
{
    const int row  = blockIdx.x;
    const int lane = threadIdx.x;
    const float a = logits[(size_t)row * CC + lane];
    const float b = logits[(size_t)row * CC + 32 + lane];
    float m = fmaxf(a, b);
    #pragma unroll
    for (int o = 16; o > 0; o >>= 1) m = fmaxf(m, __shfl_xor(m, o, 32));
    float s = __expf(a - m) + __expf(b - m);
    #pragma unroll
    for (int o = 16; o > 0; o >>= 1) s += __shfl_xor(s, o, 32);
    const float ls = __logf(s);
    out[(size_t)row * CC + lane]      = a - m - ls;
    out[(size_t)row * CC + 32 + lane] = b - m - ls;
}

// ---------------------------------------------------------------------------
// Host launch
// ---------------------------------------------------------------------------
extern "C" void kernel_launch(void* const* d_in, const int* in_sizes, int n_in,
                              void* d_out, int out_size, void* d_ws, size_t ws_size,
                              hipStream_t stream)
{
    (void)in_sizes; (void)n_in; (void)out_size; (void)ws_size;

    const float* X   = (const float*)d_in[0];
    const int*   A   = (const int*)  d_in[1];
    const float* Wn1 = (const float*)d_in[2];
    const float* bn1 = (const float*)d_in[3];
    const float* Wl1 = (const float*)d_in[4];
    const float* bl1 = (const float*)d_in[5];
    const float* Wn2 = (const float*)d_in[6];
    const float* bn2 = (const float*)d_in[7];
    const float* Wl2 = (const float*)d_in[8];
    const float* bl2 = (const float*)d_in[9];
    const float* W3  = (const float*)d_in[10];
    const float* b3  = (const float*)d_in[11];
    float* out = (float*)d_out;

    char* ws = (char*)d_ws;
    size_t off = 0;
    auto wsalloc = [&](size_t bytes) -> char* {
        char* p = ws + off;
        off = (off + bytes + 255) & ~(size_t)255;
        return p;
    };

    int*    idx    = (int*)   wsalloc((size_t)NN * KN1 * sizeof(int));
    int*    cnt1   = (int*)   wsalloc((size_t)NN * sizeof(int));
    int*    cnt2   = (int*)   wsalloc((size_t)NN * sizeof(int));
    __bf16* Xb     = (__bf16*)wsalloc((size_t)NN * FF * 2);
    __bf16* M1b    = (__bf16*)wsalloc((size_t)NN * FF * 2);
    __bf16* H1b    = (__bf16*)wsalloc((size_t)NN * FF * 2);
    __bf16* M2b    = (__bf16*)wsalloc((size_t)NN * FF * 2);
    __bf16* H2b    = (__bf16*)wsalloc((size_t)NN * FF * 2);
    __bf16* Wn1b   = (__bf16*)wsalloc((size_t)FF * FF * 2);
    __bf16* Wl1b   = (__bf16*)wsalloc((size_t)FF * FF * 2);
    __bf16* Wn2b   = (__bf16*)wsalloc((size_t)FF * FF * 2);
    __bf16* Wl2b   = (__bf16*)wsalloc((size_t)FF * FF * 2);
    __bf16* W3b    = (__bf16*)wsalloc((size_t)CC * FF * 2);
    float*  logits = (float*) wsalloc((size_t)NN * CC * sizeof(float));

    // Neighbor selection (independent of conversions).
    select_first_k_kernel<<<NN, 32, 0, stream>>>(A, idx, cnt1, cnt2);

    // bf16 conversions.
    f32_to_bf16_kernel<<<1024, 256, 0, stream>>>(X,   Xb,   NN * FF);
    f32_to_bf16_kernel<<<64,   256, 0, stream>>>(Wn1, Wn1b, FF * FF);
    f32_to_bf16_kernel<<<64,   256, 0, stream>>>(Wl1, Wl1b, FF * FF);
    f32_to_bf16_kernel<<<64,   256, 0, stream>>>(Wn2, Wn2b, FF * FF);
    f32_to_bf16_kernel<<<64,   256, 0, stream>>>(Wl2, Wl2b, FF * FF);
    f32_to_bf16_kernel<<<16,   256, 0, stream>>>(W3,  W3b,  CC * FF);

    // Layer 1: mean over first-25 neighbors, dual GEMM + leaky + gate.
    aggregate_mean_kernel<<<NN, FF, 0, stream>>>(Xb, idx, cnt1, M1b);
    dual_gemm_leaky_kernel<<<NN / 16, 256, 0, stream>>>(Xb, M1b, Wl1b, bl1, Wn1b, bn1, cnt1, H1b);

    // Layer 2: mean over first-10 neighbors (prefix of same index list).
    aggregate_mean_kernel<<<NN, FF, 0, stream>>>(H1b, idx, cnt2, M2b);
    dual_gemm_leaky_kernel<<<NN / 16, 256, 0, stream>>>(H1b, M2b, Wl2b, bl2, Wn2b, bn2, cnt2, H2b);

    // Classifier + log_softmax.
    final_gemm_kernel<<<NN / 16, 128, 0, stream>>>(H2b, W3b, b3, logits);
    log_softmax64_kernel<<<NN, 32, 0, stream>>>(logits, out);
}
